// Model_7516192768071
// MI455X (gfx1250) — matrix operand order repaired and checked by reading, not compiled, not run on hardware
//
#include <hip/hip_runtime.h>
#include <hip/hip_bf16.h>

// Problem constants (from reference):
//   B=32, H=8, CIN=256, COUT=256, L=1024, MODES=512, BH = B*H = 256
#define FNO_L      1024
#define FNO_MODES  512
#define FNO_CIN    256
#define FNO_COUT   256
#define FNO_BH     256

typedef _Float16 v16h __attribute__((ext_vector_type(16)));
typedef float    v8f  __attribute__((ext_vector_type(8)));
typedef unsigned v8u  __attribute__((ext_vector_type(8)));

// ---------------------------------------------------------------------------
// Kernel 1: forward 1024-pt real FFT per (b,h,i) row; write modes 0..511 as
// f16 into a[m][bh][i] (contiguous in i -> feeds WMMA B-operand directly).
// ---------------------------------------------------------------------------
__global__ void __launch_bounds__(256)
fno_fft_fwd(const float* __restrict__ x,
            _Float16* __restrict__ aRe,
            _Float16* __restrict__ aIm) {
    __shared__ float re[FNO_L];
    __shared__ float im[FNO_L];
    const int row = blockIdx.x;                 // (b*H + h)*CIN + i
    const int t   = threadIdx.x;                // 0..255
    const float* px = x + (size_t)row * FNO_L;

    // bit-reversed load (10-bit reversal), imag = 0
    #pragma unroll
    for (int j = 0; j < 4; ++j) {
        int n  = t + j * 256;
        int br = (int)(__brev((unsigned)n) >> 22);
        re[br] = px[n];
        im[br] = 0.0f;
    }
    __syncthreads();

    // radix-2 DIT, twiddle exp(-2*pi*i * pos/(2*len))
    for (int len = 1; len < FNO_L; len <<= 1) {
        #pragma unroll 2
        for (int j = t; j < FNO_L / 2; j += 256) {
            int blk = j / len;
            int pos = j - blk * len;
            int i0  = blk * (len << 1) + pos;
            int i1  = i0 + len;
            float ang = -3.14159265358979323846f * (float)pos / (float)len;
            float s, c;
            __sincosf(ang, &s, &c);
            float xr = re[i1], xi = im[i1];
            float tr = xr * c - xi * s;
            float ti = xr * s + xi * c;
            float ur = re[i0], ui = im[i0];
            re[i0] = ur + tr;  im[i0] = ui + ti;
            re[i1] = ur - tr;  im[i1] = ui - ti;
        }
        __syncthreads();
    }

    const int bh = row >> 8;      // CIN == 256
    const int ci = row & 255;
    #pragma unroll
    for (int j = 0; j < 2; ++j) {
        int m = t + j * 256;      // mode 0..511
        size_t off = (((size_t)m * FNO_BH + bh) * FNO_CIN) + ci;
        aRe[off] = (_Float16)re[m];
        aIm[off] = (_Float16)im[m];
    }
}

// ---------------------------------------------------------------------------
// Kernel 2: pack weights f32 -> f16 in WMMA A-operand register layout.
// A-operand (16-bit 16x32, wave32): lane L: M = L%16, half = L/16;
// element e: K = half*16 + e (contiguous k order chosen here; the B loads use the same).
// Tile id = ((m*16 + oTile)*8 + kTile), 512 f16 per tile (lane*16 + e).
// ---------------------------------------------------------------------------
__global__ void __launch_bounds__(256)
fno_pack_w(const float* __restrict__ wre,
           const float* __restrict__ wim,
           _Float16* __restrict__ wpR,
           _Float16* __restrict__ wpI) {
    const int m     = blockIdx.x;   // 0..511
    const int oTile = blockIdx.y;   // 0..15
    const int kTile = blockIdx.z;   // 0..7
    const int t     = threadIdx.x;  // 0..255
    const size_t tileBase = (((size_t)m * 16 + oTile) * 8 + kTile) * 512;

    #pragma unroll
    for (int j = 0; j < 2; ++j) {
        int idx  = t + j * 256;     // 0..511 within tile
        int L    = idx >> 4;        // lane
        int e    = idx & 15;        // element within v16h
        int half = L >> 4;
        int o    = oTile * 16 + (L & 15);
        int p    = e >> 1;
        int kk   = half * 16 + e; (void)p;
        int i    = kTile * 32 + kk;
        size_t src = ((size_t)i * FNO_COUT + o) * FNO_MODES + m;
        wpR[tileBase + idx] = (_Float16)wre[src];
        wpI[tileBase + idx] = (_Float16)wim[src];
    }
}

// ---------------------------------------------------------------------------
// Kernel 3: per-mode complex GEMM  C[o][r] = sum_i Wt[o][i] * At[i][r]
// using v_wmma_f32_16x16x32_f16, 2x2 register blocking per wave:
// each wave owns a 32x32 output patch -> 16 WMMAs per 8 operand loads.
// __launch_bounds__(256,1) lifts the VGPR cap so the ~170-reg working set
// (8 v8f accumulators + 8 v16h operands) stays in registers (no spills).
// Grid: (MODES, 4, 2); block = 256 threads = 8 waves = 4 oPairs x 2 rPairs.
// All indices are 32-bit (arrays < 2^31 elements) to trim u64 address chains.
// ---------------------------------------------------------------------------
__global__ void __launch_bounds__(256, 1)
fno_spec_gemm(const _Float16* __restrict__ wpR,
              const _Float16* __restrict__ wpI,
              const _Float16* __restrict__ aR,
              const _Float16* __restrict__ aI,
              float* __restrict__ oR,
              float* __restrict__ oI) {
    const int m    = blockIdx.x;               // mode
    const int wave = threadIdx.x >> 5;         // 0..7
    const int lane = threadIdx.x & 31;
    const int oPair = blockIdx.z * 4 + (wave & 3);   // 0..7
    const int rPair = blockIdx.y * 2 + (wave >> 2);  // 0..7
    const int oT0 = oPair * 2, oT1 = oT0 + 1;        // COUT tiles
    const int rT0 = rPair * 2, rT1 = rT0 + 1;        // BH tiles

    const int laneHalf = lane >> 4;
    const int laneM    = lane & 15;
    const int r0 = rT0 * 16 + laneM;           // B-operand column (tile 0)
    const int r1 = rT1 * 16 + laneM;           // B-operand column (tile 1)

    v8f aR00 = {}, aR01 = {}, aR10 = {}, aR11 = {};
    v8f aI00 = {}, aI01 = {}, aI10 = {}, aI11 = {};

    const int aBase0 = (m * FNO_BH + r0) * FNO_CIN + laneHalf * 16;
    const int aBase1 = (m * FNO_BH + r1) * FNO_CIN + laneHalf * 16;
    const int wBase0 = ((m * 16 + oT0) * 8) * 512 + lane * 16;
    const int wBase1 = ((m * 16 + oT1) * 8) * 512 + lane * 16;

    #pragma unroll
    for (int k = 0; k < 8; ++k) {              // K = 8 * 32 = 256
        const int w0 = wBase0 + k * 512;
        const int w1 = wBase1 + k * 512;
        const int a0 = aBase0 + k * 32;
        const int a1 = aBase1 + k * 32;

        v16h wr0 = *(const v16h*)(wpR + w0);
        v16h wi0 = *(const v16h*)(wpI + w0);
        v16h wr1 = *(const v16h*)(wpR + w1);
        v16h wi1 = *(const v16h*)(wpI + w1);
        v16h ar0 = *(const v16h*)(aR + a0);
        v16h ai0 = *(const v16h*)(aI + a0);
        v16h ar1 = *(const v16h*)(aR + a1);
        v16h ai1 = *(const v16h*)(aI + a1);

        // -Wi via sign-bit xor on packed halves (f16 WMMA has no A/B negate)
        v16h wn0 = __builtin_bit_cast(v16h, __builtin_bit_cast(v8u, wi0) ^ 0x80008000u);
        v16h wn1 = __builtin_bit_cast(v16h, __builtin_bit_cast(v8u, wi1) ^ 0x80008000u);

        // (o0, r0)
        aR00 = __builtin_amdgcn_wmma_f32_16x16x32_f16(false, wr0, false, ar0, (short)0, aR00, false, false);
        aR00 = __builtin_amdgcn_wmma_f32_16x16x32_f16(false, wn0, false, ai0, (short)0, aR00, false, false);
        aI00 = __builtin_amdgcn_wmma_f32_16x16x32_f16(false, wr0, false, ai0, (short)0, aI00, false, false);
        aI00 = __builtin_amdgcn_wmma_f32_16x16x32_f16(false, wi0, false, ar0, (short)0, aI00, false, false);
        // (o0, r1)
        aR01 = __builtin_amdgcn_wmma_f32_16x16x32_f16(false, wr0, false, ar1, (short)0, aR01, false, false);
        aR01 = __builtin_amdgcn_wmma_f32_16x16x32_f16(false, wn0, false, ai1, (short)0, aR01, false, false);
        aI01 = __builtin_amdgcn_wmma_f32_16x16x32_f16(false, wr0, false, ai1, (short)0, aI01, false, false);
        aI01 = __builtin_amdgcn_wmma_f32_16x16x32_f16(false, wi0, false, ar1, (short)0, aI01, false, false);
        // (o1, r0)
        aR10 = __builtin_amdgcn_wmma_f32_16x16x32_f16(false, wr1, false, ar0, (short)0, aR10, false, false);
        aR10 = __builtin_amdgcn_wmma_f32_16x16x32_f16(false, wn1, false, ai0, (short)0, aR10, false, false);
        aI10 = __builtin_amdgcn_wmma_f32_16x16x32_f16(false, wr1, false, ai0, (short)0, aI10, false, false);
        aI10 = __builtin_amdgcn_wmma_f32_16x16x32_f16(false, wi1, false, ar0, (short)0, aI10, false, false);
        // (o1, r1)
        aR11 = __builtin_amdgcn_wmma_f32_16x16x32_f16(false, wr1, false, ar1, (short)0, aR11, false, false);
        aR11 = __builtin_amdgcn_wmma_f32_16x16x32_f16(false, wn1, false, ai1, (short)0, aR11, false, false);
        aI11 = __builtin_amdgcn_wmma_f32_16x16x32_f16(false, wr1, false, ai1, (short)0, aI11, false, false);
        aI11 = __builtin_amdgcn_wmma_f32_16x16x32_f16(false, wi1, false, ar1, (short)0, aI11, false, false);
    }

    // C/D layout: lane: N = lane%16 (-> r), VGPR vg: M = vg + (lane/16)*8 (-> o)
    #pragma unroll
    for (int vg = 0; vg < 8; ++vg) {
        int o0 = oT0 * 16 + vg + laneHalf * 8;
        int o1 = oT1 * 16 + vg + laneHalf * 8;
        int off00 = ((r0 * FNO_COUT + o0) * FNO_MODES) + m;
        int off01 = ((r1 * FNO_COUT + o0) * FNO_MODES) + m;
        int off10 = ((r0 * FNO_COUT + o1) * FNO_MODES) + m;
        int off11 = ((r1 * FNO_COUT + o1) * FNO_MODES) + m;
        oR[off00] = aR00[vg];  oI[off00] = aI00[vg];
        oR[off01] = aR01[vg];  oI[off01] = aI01[vg];
        oR[off10] = aR10[vg];  oI[off10] = aI10[vg];
        oR[off11] = aR11[vg];  oI[off11] = aI11[vg];
    }
}

// ---------------------------------------------------------------------------
// Kernel 4: inverse rfft per (b,h,o) row: build conj of the Hermitian
// spectrum (mode 512 = 0 pad), forward FFT it, out = Re(.)/1024.
// ---------------------------------------------------------------------------
__global__ void __launch_bounds__(256)
fno_ifft(const float* __restrict__ oR,
         const float* __restrict__ oI,
         float* __restrict__ out) {
    __shared__ float re[FNO_L];
    __shared__ float im[FNO_L];
    const int row = blockIdx.x;                // bh*COUT + o
    const int t   = threadIdx.x;
    const float* pR = oR + (size_t)row * FNO_MODES;
    const float* pI = oI + (size_t)row * FNO_MODES;

    #pragma unroll
    for (int j = 0; j < 4; ++j) {
        int k = t + j * 256;
        float xr, xi;
        if (k < FNO_MODES)      { xr = pR[k];        xi = -pI[k]; }        // conj(X[k])
        else if (k == FNO_MODES) { xr = 0.0f;        xi = 0.0f;   }        // padded mode
        else { int kk = FNO_L - k; xr = pR[kk];      xi = pI[kk]; }        // conj(conj(X)) = X
        int br = (int)(__brev((unsigned)k) >> 22);
        re[br] = xr;
        im[br] = xi;
    }
    __syncthreads();

    for (int len = 1; len < FNO_L; len <<= 1) {
        #pragma unroll 2
        for (int j = t; j < FNO_L / 2; j += 256) {
            int blk = j / len;
            int pos = j - blk * len;
            int i0  = blk * (len << 1) + pos;
            int i1  = i0 + len;
            float ang = -3.14159265358979323846f * (float)pos / (float)len;
            float s, c;
            __sincosf(ang, &s, &c);
            float xr = re[i1], xi = im[i1];
            float tr = xr * c - xi * s;
            float ti = xr * s + xi * c;
            float ur = re[i0], ui = im[i0];
            re[i0] = ur + tr;  im[i0] = ui + ti;
            re[i1] = ur - tr;  im[i1] = ui - ti;
        }
        __syncthreads();
    }

    float* po = out + (size_t)row * FNO_L;
    const float invN = 1.0f / (float)FNO_L;
    #pragma unroll
    for (int j = 0; j < 4; ++j) {
        int n = t + j * 256;
        po[n] = re[n] * invN;     // FFT(conj(X)) = conj(N * irfft); output is real
    }
}

// ---------------------------------------------------------------------------
// Launch
// ---------------------------------------------------------------------------
extern "C" void kernel_launch(void* const* d_in, const int* in_sizes, int n_in,
                              void* d_out, int out_size, void* d_ws, size_t ws_size,
                              hipStream_t stream) {
    (void)in_sizes; (void)n_in; (void)out_size; (void)ws_size;

    const float* x      = (const float*)d_in[0];   // (32,8,256,1024) f32
    const float* w_real = (const float*)d_in[1];   // (256,256,512) f32
    const float* w_imag = (const float*)d_in[2];   // (256,256,512) f32
    float* out = (float*)d_out;                    // (32,8,256,1024) f32

    char* ws = (char*)d_ws;
    const size_t MB = 1024u * 1024u;
    _Float16* aRe = (_Float16*)(ws);               //  64 MB: a[m][bh][i] f16
    _Float16* aIm = (_Float16*)(ws + 64 * MB);     //  64 MB
    _Float16* wpR = (_Float16*)(ws + 128 * MB);    //  64 MB: packed W tiles
    _Float16* wpI = (_Float16*)(ws + 192 * MB);    //  64 MB
    float*    oRe = (float*)   (ws + 256 * MB);    // 128 MB: O[bh][o][m] f32
    float*    oIm = (float*)   (ws + 384 * MB);    // 128 MB

    // 1) forward rfft of every (b,h,i) row
    fno_fft_fwd<<<dim3(32 * 8 * FNO_CIN), dim3(256), 0, stream>>>(x, aRe, aIm);

    // 2) weight conversion + WMMA A-operand swizzle
    fno_pack_w<<<dim3(FNO_MODES, 16, 8), dim3(256), 0, stream>>>(w_real, w_imag, wpR, wpI);

    // 3) 512 per-mode complex GEMMs on the WMMA pipes (2x2 register blocking)
    fno_spec_gemm<<<dim3(FNO_MODES, 4, 2), dim3(256), 0, stream>>>(
        wpR, wpI, aRe, aIm, oRe, oIm);

    // 4) inverse rfft of every (b,h,o) row
    fno_ifft<<<dim3(32 * 8 * FNO_COUT), dim3(256), 0, stream>>>(oRe, oIm, out);
}
